// CapsNet_78537771975026
// MI455X (gfx1250) — compile-verified, matrix operand order checked by reading, not run on hardware
//
#include <hip/hip_runtime.h>
#include <hip/hip_bf16.h>
#include <math.h>

typedef __attribute__((ext_vector_type(16))) __bf16 bf16x16;
typedef __attribute__((ext_vector_type(8)))  float  f32x8;

#define DEVINL static __device__ __forceinline__

DEVINL unsigned short f2bf_bits(float f) {
    unsigned int u = __float_as_uint(f);
    u += 0x7FFFu + ((u >> 16) & 1u);   // round-to-nearest-even
    return (unsigned short)(u >> 16);
}
DEVINL float bf2f(unsigned short b) { return __uint_as_float(((unsigned int)b) << 16); }

DEVINL f32x8 wmma_bf16(bf16x16 a, bf16x16 b, f32x8 c) {
    return __builtin_amdgcn_wmma_f32_16x16x32_bf16(false, a, false, b, (short)0, c, false, false);
}

// Slot->memory K map used by BOTH operands (any per-32-block permutation is
// legal as long as A and B agree): slot (e, half) <- k_local = e + 16*half.
// This makes every A fragment one contiguous 32-byte load.

// ---------------------------------------------------------------------------
// conv1 (1->256, 9x9, valid) + ReLU; output NHWC bf16 feat[b][y][x][cin].
// cin fastest -> wave shares one input window (broadcast loads), coalesced
// NHWC stores.
// ---------------------------------------------------------------------------
__global__ void conv1_relu_bf16(const float* __restrict__ x, const float* __restrict__ w,
                                const float* __restrict__ bias, unsigned short* __restrict__ feat) {
    long idx = (long)blockIdx.x * blockDim.x + threadIdx.x;
    const long total = 512L * 20 * 20 * 256;
    if (idx >= total) return;
    int co = (int)(idx & 255); long t = idx >> 8;
    int ox = (int)(t % 20);
    int oy = (int)((t / 20) % 20);
    int b  = (int)(t / 400);
    const float* xb = x + (long)b * 784;
    const float* wc = w + (long)co * 81;
    float acc = bias[co];
#pragma unroll
    for (int ky = 0; ky < 9; ++ky) {
        const float* xr = xb + (oy + ky) * 28 + ox;
        const float* wr = wc + ky * 9;
#pragma unroll
        for (int kx = 0; kx < 9; ++kx) acc = fmaf(xr[kx], wr[kx], acc);
    }
    feat[idx] = f2bf_bits(fmaxf(acc, 0.f));
}

// ---------------------------------------------------------------------------
// Pack [N][K] f32 weight matrix into bf16 WMMA B-fragment layout using the
// remapped slots: lane = half*16 + n%16, element e -> k = kblk*32 + e + 16*half.
// ---------------------------------------------------------------------------
__global__ void pack_b_frag(const float* __restrict__ W, unsigned short* __restrict__ out,
                            int N, int K) {
    int nkblk = K >> 5;
    long total = (long)(N >> 4) * nkblk * 32 * 16;
    for (long idx = (long)blockIdx.x * blockDim.x + threadIdx.x; idx < total;
         idx += (long)gridDim.x * blockDim.x) {
        int e = (int)(idx & 15);
        long t = idx >> 4;
        int lane = (int)(t & 31); t >>= 5;
        int kblk  = (int)(t % nkblk);
        int ntile = (int)(t / nkblk);
        int half = lane >> 4, nl = lane & 15;
        int k = kblk * 32 + e + 16 * half;
        int n = ntile * 16 + nl;
        out[idx] = f2bf_bits(W[(long)n * K + k]);
    }
}

// Same, but for pconv weights with permuted K-order k' = (ky*9+kx)*256 + cin.
// Source pconv_w is [co][cin][ky][kx] f32.
__global__ void pack_b_frag_pconv(const float* __restrict__ W, unsigned short* __restrict__ out) {
    const int nkblk = 648;                    // K = 20736
    long total = 16L * nkblk * 32 * 16;       // N=256 -> 16 ntiles
    for (long idx = (long)blockIdx.x * blockDim.x + threadIdx.x; idx < total;
         idx += (long)gridDim.x * blockDim.x) {
        int e = (int)(idx & 15);
        long t = idx >> 4;
        int lane = (int)(t & 31); t >>= 5;
        int kblk  = (int)(t % nkblk);
        int ntile = (int)(t / nkblk);
        int half = lane >> 4, nl = lane & 15;
        int k = kblk * 32 + e + 16 * half;
        int tap = k >> 8;                     // ky*9 + kx
        int cin = k & 255;
        int n = ntile * 16 + nl;              // co
        out[idx] = f2bf_bits(W[(long)n * 20736 + (long)cin * 81 + tap]);
    }
}

// ---------------------------------------------------------------------------
// PrimaryCaps conv as implicit-im2col WMMA GEMM, 32x64 tile per wave.
// M = 512*36 (b,oy,ox), N = 256 (co), K = 81 taps * 256 cin = 20736,
// K-order (tap, cin): each A fragment is ONE contiguous 32B load from NHWC
// feat; per K-block: 8 WMMA : 6 b128-class loads.
// Epilogue scatters into u_pre[b][i=cap*36+s][d].
// ---------------------------------------------------------------------------
__global__ void pconv_wmma(const unsigned short* __restrict__ feat,
                           const unsigned short* __restrict__ Bp,
                           const float* __restrict__ bias,
                           float* __restrict__ u_pre) {
    const int lane = threadIdx.x;
    const int half = lane >> 4, ml = lane & 15;
    const int mt0 = blockIdx.x * 2;           // two m-tiles
    const int nt0 = blockIdx.y * 4;           // four n-tiles
    const int nkblk = 648;

    long pa[2];
#pragma unroll
    for (int i = 0; i < 2; ++i) {
        int m = (mt0 + i) * 16 + ml;
        int b = m / 36, s = m % 36;
        int oy = s / 6, ox = s % 6;
        // NHWC window origin, with the half-lane K offset folded in
        pa[i] = ((long)(b * 20 + oy * 2) * 20 + ox * 2) * 256 + 16 * half;
    }
    const unsigned short* bfr[4];
#pragma unroll
    for (int j = 0; j < 4; ++j)
        bfr[j] = Bp + ((long)(nt0 + j) * nkblk * 32 + lane) * 16;

    f32x8 accs[2][4];
#pragma unroll
    for (int i = 0; i < 2; ++i)
#pragma unroll
        for (int j = 0; j < 4; ++j) accs[i][j] = (f32x8){0.f,0.f,0.f,0.f,0.f,0.f,0.f,0.f};

    for (int kb = 0; kb < nkblk; ++kb) {
        int kk = kb * 32;
        int tap = kk >> 8;                    // wave-uniform -> SALU
        int c0  = kk & 255;
        int ky = tap / 9, kx = tap - ky * 9;
        long off = ((long)(ky * 20 + kx)) * 256 + c0;

        bf16x16 a0 = *(const bf16x16*)(feat + pa[0] + off);
        bf16x16 a1 = *(const bf16x16*)(feat + pa[1] + off);
#pragma unroll
        for (int j = 0; j < 4; ++j) {
            bf16x16 bv = *(const bf16x16*)(bfr[j] + (long)kb * 512);
            accs[0][j] = wmma_bf16(a0, bv, accs[0][j]);
            accs[1][j] = wmma_bf16(a1, bv, accs[1][j]);
        }
        if (kb + 4 < nkblk) {
            __builtin_prefetch(bfr[0] + (long)(kb + 4) * 512, 0, 0);
            __builtin_prefetch(bfr[2] + (long)(kb + 4) * 512, 0, 0);
        }
    }

#pragma unroll
    for (int i = 0; i < 2; ++i) {
#pragma unroll
        for (int j = 0; j < 4; ++j) {
            int co = (nt0 + j) * 16 + ml;
            float bco = bias[co];
#pragma unroll
            for (int r = 0; r < 8; ++r) {
                int mm = (mt0 + i) * 16 + r + 8 * half;
                int bb = mm / 36, ss = mm % 36;
                u_pre[((long)bb * 1152 + (co >> 3) * 36 + ss) * 8 + (co & 7)] =
                    accs[i][j][r] + bco;
            }
        }
    }
}

// ---------------------------------------------------------------------------
// squash over the 8-dim capsule axis; store bf16 u for the u_hat einsum.
// ---------------------------------------------------------------------------
__global__ void squash_u_kernel(const float* __restrict__ u_pre, unsigned short* __restrict__ u_bf) {
    long idx = (long)blockIdx.x * blockDim.x + threadIdx.x;  // (b,i)
    if (idx >= 512L * 1152) return;
    const float* s = u_pre + idx * 8;
    float v[8]; float ns = 0.f;
#pragma unroll
    for (int d = 0; d < 8; ++d) { v[d] = s[d]; ns = fmaf(v[d], v[d], ns); }
    float nrm = sqrtf(ns);
    float f = ns / (1.f + ns) / (nrm + 1e-8f);
    unsigned short* o = u_bf + idx * 8;
#pragma unroll
    for (int d = 0; d < 8; ++d) o[d] = f2bf_bits(v[d] * f);
}

// ---------------------------------------------------------------------------
// u_hat[b,i,c,d] = sum_k W[i,c,d,k] * u[b,i,k]  (K=8 -> VALU), bf16 out.
// ---------------------------------------------------------------------------
__global__ void uhat_kernel(const unsigned short* __restrict__ u_bf, const float* __restrict__ W,
                            unsigned short* __restrict__ u_hat) {
    long idx = (long)blockIdx.x * blockDim.x + threadIdx.x;  // (b,i,c)
    if (idx >= 512L * 1152 * 10) return;
    int c = (int)(idx % 10); long t = idx / 10;
    int i = (int)(t % 1152);
    float u[8];
    const unsigned short* up = u_bf + t * 8;
#pragma unroll
    for (int k = 0; k < 8; ++k) u[k] = bf2f(up[k]);
    const float* wp = W + ((long)i * 10 + c) * 128;
    unsigned short* o = u_hat + idx * 16;
#pragma unroll
    for (int d = 0; d < 16; ++d) {
        float acc = 0.f;
#pragma unroll
        for (int k = 0; k < 8; ++k) acc = fmaf(wp[d * 8 + k], u[k], acc);
        o[d] = f2bf_bits(acc);
    }
}

// ---------------------------------------------------------------------------
// Routing iteration kernels
// ---------------------------------------------------------------------------
__global__ void routing_softmax(const float* __restrict__ b_logits, float* __restrict__ c_coef) {
    long idx = (long)blockIdx.x * blockDim.x + threadIdx.x;  // (b,i)
    if (idx >= 512L * 1152) return;
    const float* bl = b_logits + idx * 10;
    float mx = bl[0];
#pragma unroll
    for (int c = 1; c < 10; ++c) mx = fmaxf(mx, bl[c]);
    float e[10], sum = 0.f;
#pragma unroll
    for (int c = 0; c < 10; ++c) { e[c] = __expf(bl[c] - mx); sum += e[c]; }
    float inv = 1.f / sum;
    float* o = c_coef + idx * 10;
#pragma unroll
    for (int c = 0; c < 10; ++c) o[c] = e[c] * inv;
}

// one block per batch element, 160 threads = (c,d); s = sum_i c*u_hat; v = squash(s)
__global__ void routing_s_v(const float* __restrict__ c_coef, const unsigned short* __restrict__ u_hat,
                            float* __restrict__ v, float* __restrict__ v_extra) {
    int b = blockIdx.x;
    int t = threadIdx.x;            // 0..159
    int c = t >> 4, d = t & 15;
    const float* cc = c_coef + (long)b * 11520 + c;
    const unsigned short* uh = u_hat + (long)b * 184320 + (long)c * 16 + d;
    float acc = 0.f;
    for (int i = 0; i < 1152; ++i)
        acc = fmaf(cc[(long)i * 10], bf2f(uh[(long)i * 160]), acc);
    __shared__ float sv[160];
    sv[t] = acc;
    __syncthreads();
    float ns = 0.f;
#pragma unroll
    for (int dd = 0; dd < 16; ++dd) { float xx = sv[c * 16 + dd]; ns = fmaf(xx, xx, ns); }
    float nrm = sqrtf(ns);
    float f = ns / (1.f + ns) / (nrm + 1e-8f);
    float val = f * acc;
    v[(long)b * 160 + t] = val;
    if (v_extra) v_extra[(long)b * 160 + t] = val;
}

__global__ void routing_agree(const unsigned short* __restrict__ u_hat, const float* __restrict__ v,
                              float* __restrict__ b_logits) {
    long idx = (long)blockIdx.x * blockDim.x + threadIdx.x;  // (b,i,c)
    if (idx >= 512L * 1152 * 10) return;
    int c = (int)(idx % 10); long t = idx / 10;
    int b = (int)(t / 1152);
    const unsigned short* uh = u_hat + idx * 16;
    const float* vp = v + (long)b * 160 + c * 16;
    float acc = 0.f;
#pragma unroll
    for (int d = 0; d < 16; ++d) acc = fmaf(bf2f(uh[d]), vp[d], acc);
    b_logits[idx] += acc;
}

__global__ void v_length_kernel(const float* __restrict__ v, float* __restrict__ out) {
    int idx = blockIdx.x * blockDim.x + threadIdx.x;  // (b,c)
    if (idx >= 5120) return;
    const float* vp = v + (long)idx * 16;
    float ns = 0.f;
#pragma unroll
    for (int d = 0; d < 16; ++d) ns = fmaf(vp[d], vp[d], ns);
    out[idx] = sqrtf(ns);
}

__global__ void mask_kernel(const float* __restrict__ v, const int* __restrict__ labels,
                            unsigned short* __restrict__ masked) {
    int idx = blockIdx.x * blockDim.x + threadIdx.x;  // (b,t) 512*160
    if (idx >= 512 * 160) return;
    int t = idx % 160; int b = idx / 160;
    float val = (labels[b] == (t >> 4)) ? v[idx] : 0.f;
    masked[idx] = f2bf_bits(val);
}

// ---------------------------------------------------------------------------
// Generic bf16 WMMA GEMM: out[M,N] = act(A[M,K] @ Bpacked + bias).
// One wave per 16x16 tile; A fragment = single contiguous 32B load.
// act: 0 none, 1 relu, 2 sigmoid.
// ---------------------------------------------------------------------------
__global__ void gemm_bf16_wmma(const unsigned short* __restrict__ A,
                               const unsigned short* __restrict__ Bp,
                               const float* __restrict__ bias,
                               float* __restrict__ outF, unsigned short* __restrict__ outBF,
                               int N, int K, int act) {
    const int lane = threadIdx.x;
    const int half = lane >> 4, ml = lane & 15;
    const int mtile = blockIdx.x, ntile = blockIdx.y;
    const int nkblk = K >> 5;
    const unsigned short* arow = A + (long)(mtile * 16 + ml) * K + 16 * half;
    const unsigned short* bfrag = Bp + ((long)ntile * nkblk * 32 + lane) * 16;
    f32x8 acc = {0.f, 0.f, 0.f, 0.f, 0.f, 0.f, 0.f, 0.f};
    for (int kb = 0; kb < nkblk; ++kb) {
        bf16x16 av = *(const bf16x16*)(arow + kb * 32);
        bf16x16 bv = *(const bf16x16*)(bfrag + (long)kb * 512);
        acc = wmma_bf16(av, bv, acc);
        if (kb + 4 < nkblk) __builtin_prefetch(bfrag + (long)(kb + 4) * 512, 0, 0);
    }
    int n = ntile * 16 + ml;
    float bn = bias[n];
#pragma unroll
    for (int r = 0; r < 8; ++r) {
        int m = mtile * 16 + r + 8 * half;
        float val = acc[r] + bn;
        if (act == 1) val = fmaxf(val, 0.f);
        else if (act == 2) val = 1.f / (1.f + __expf(-val));
        long o = (long)m * N + n;
        if (outF)  outF[o] = val;
        if (outBF) outBF[o] = f2bf_bits(val);
    }
}

// ---------------------------------------------------------------------------
extern "C" void kernel_launch(void* const* d_in, const int* in_sizes, int n_in,
                              void* d_out, int out_size, void* d_ws, size_t ws_size,
                              hipStream_t stream) {
    const float* x       = (const float*)d_in[0];
    const int*   labels  = (const int*)d_in[1];
    const float* conv1_w = (const float*)d_in[2];
    const float* conv1_b = (const float*)d_in[3];
    const float* pconv_w = (const float*)d_in[4];
    const float* pconv_b = (const float*)d_in[5];
    const float* Wroute  = (const float*)d_in[6];
    const float* fc1_w   = (const float*)d_in[7];
    const float* fc1_b   = (const float*)d_in[8];
    const float* fc2_w   = (const float*)d_in[9];
    const float* fc2_b   = (const float*)d_in[10];
    const float* fc3_w   = (const float*)d_in[11];
    const float* fc3_b   = (const float*)d_in[12];
    (void)in_sizes; (void)n_in; (void)out_size; (void)ws_size;
    float* out = (float*)d_out;

    char* ws = (char*)d_ws;
    size_t off = 0;
    auto take = [&](size_t bytes) -> char* {
        size_t o = (off + 255) & ~(size_t)255;
        off = o + bytes;
        return ws + o;
    };
    unsigned short* feat    = (unsigned short*)take(512L * 400 * 256 * 2);   // NHWC, 105 MB
    unsigned short* wpP     = (unsigned short*)take(648L * 16 * 512 * 2);    // 10.6 MB
    float*          u_pre   = (float*)take(512L * 1152 * 8 * 4);             // 18.9 MB
    unsigned short* u_bf    = (unsigned short*)take(512L * 1152 * 8 * 2);    // 9.4 MB
    unsigned short* u_hat   = (unsigned short*)take(512L * 1152 * 160 * 2);  // 189 MB
    float*          blog    = (float*)take(512L * 1152 * 10 * 4);            // 23.6 MB
    float*          ccoef   = (float*)take(512L * 1152 * 10 * 4);            // 23.6 MB
    float*          vbuf    = (float*)take(512L * 160 * 4);
    unsigned short* maskedb = (unsigned short*)take(512L * 160 * 2);
    unsigned short* h1      = (unsigned short*)take(512L * 512 * 2);
    unsigned short* h2      = (unsigned short*)take(512L * 1024 * 2);
    unsigned short* wp1     = (unsigned short*)take(5L * 32 * 512 * 2);   // K=160,N=512
    unsigned short* wp2     = (unsigned short*)take(16L * 64 * 512 * 2);  // K=512,N=1024
    unsigned short* wp3     = (unsigned short*)take(32L * 49 * 512 * 2);  // K=1024,N=784

    // conv1 + relu -> bf16 NHWC feat
    {
        long total = 512L * 400 * 256;
        conv1_relu_bf16<<<(int)((total + 255) / 256), 256, 0, stream>>>(x, conv1_w, conv1_b, feat);
    }
    // pack GEMM B operands into WMMA fragment layout
    pack_b_frag_pconv<<<4096, 256, 0, stream>>>(pconv_w, wpP);
    pack_b_frag<<<320,  256, 0, stream>>>(fc1_w, wp1, 512, 160);
    pack_b_frag<<<2048, 256, 0, stream>>>(fc2_w, wp2, 1024, 512);
    pack_b_frag<<<3136, 256, 0, stream>>>(fc3_w, wp3, 784, 1024);
    // primary caps conv (implicit im2col WMMA GEMM, 32x64/wave) -> u_pre
    pconv_wmma<<<dim3(576, 4), 32, 0, stream>>>(feat, wpP, pconv_b, u_pre);
    // squash -> u (bf16)
    squash_u_kernel<<<(512 * 1152 + 255) / 256, 256, 0, stream>>>(u_pre, u_bf);
    // u_hat (bf16)
    {
        long total = 512L * 1152 * 10;
        uhat_kernel<<<(int)((total + 255) / 256), 256, 0, stream>>>(u_bf, Wroute, u_hat);
    }
    // dynamic routing, 3 iterations; b_logits zeroed every call (deterministic)
    hipMemsetAsync(blog, 0, 512L * 1152 * 10 * 4, stream);
    for (int it = 0; it < 3; ++it) {
        long bi = 512L * 1152;
        routing_softmax<<<(int)((bi + 255) / 256), 256, 0, stream>>>(blog, ccoef);
        routing_s_v<<<512, 160, 0, stream>>>(ccoef, u_hat, vbuf,
                                             (it == 2) ? out + 5120 + 512L * 784 : nullptr);
        if (it < 2) {
            long bic = 512L * 1152 * 10;
            routing_agree<<<(int)((bic + 255) / 256), 256, 0, stream>>>(u_hat, vbuf, blog);
        }
    }
    // outputs: v_lengths | reconstruction | v
    v_length_kernel<<<(5120 + 255) / 256, 256, 0, stream>>>(vbuf, out);
    // decoder
    mask_kernel<<<(512 * 160 + 255) / 256, 256, 0, stream>>>(vbuf, labels, maskedb);
    gemm_bf16_wmma<<<dim3(32, 32), 32, 0, stream>>>(maskedb, wp1, fc1_b, nullptr, h1, 512, 160, 1);
    gemm_bf16_wmma<<<dim3(32, 64), 32, 0, stream>>>(h1, wp2, fc2_b, nullptr, h2, 1024, 512, 1);
    gemm_bf16_wmma<<<dim3(32, 49), 32, 0, stream>>>(h2, wp3, fc3_b, out + 5120, nullptr, 784, 1024, 2);
}
